// MultiHeadAttention_33268816675348
// MI455X (gfx1250) — matrix operand order repaired and checked by reading, not compiled, not run on hardware
//
#include <hip/hip_runtime.h>
#include <cstdint>

// ---------- types ----------
typedef __attribute__((ext_vector_type(16))) _Float16 v16h;
typedef __attribute__((ext_vector_type(8)))  _Float16 v8h;
typedef __attribute__((ext_vector_type(8)))  float    v8f;
typedef __attribute__((ext_vector_type(4)))  unsigned int u32x4;
typedef __attribute__((ext_vector_type(8)))  int      i32x8;
typedef __attribute__((ext_vector_type(4)))  int      i32x4;

#define EMBED 512
#define HEADS 8
#define HD    64
#define SEQ   2048
#define NB    4
#define PIT   72   // f16 LDS row pitch (64 + 8 pad), 144B = 16B aligned rows

// ---------- WMMA helpers ----------
__device__ __forceinline__ v8f wmma_f16(v16h a, v16h b, v8f c) {
  return __builtin_amdgcn_wmma_f32_16x16x32_f16(false, a, false, b, (short)0, c,
                                                false, false);
}

// A fragment: 16x32 f16. lane m = lane%16 picks row; half hl picks k-chunk
// interleave: k = {hl*8..hl*8+7} and {16+hl*8..16+hl*8+7} per ISA A layout.
__device__ __forceinline__ v16h fragA(const _Float16* base, int hl, int koff) {
  v8h x = *(const v8h*)(base + koff + hl * 8);
  v8h y = *(const v8h*)(base + koff + 16 + hl * 8);
  return __builtin_shufflevector(x, y, 0,1,2,3,4,5,6,7,8,9,10,11,12,13,14,15);
}

// B fragment: 32x16 f16. lane n = lane%16 picks column; half hl picks k
// chunks {hl*8..hl*8+7} and {16+hl*8..16+hl*8+7}, matching fragA.
__device__ __forceinline__ v16h fragB(const _Float16* base, int hl, int koff) {
  v8h x = *(const v8h*)(base + koff + hl * 8);
  v8h y = *(const v8h*)(base + koff + 16 + hl * 8);
  return __builtin_shufflevector(x, y, 0,1,2,3,4,5,6,7,8,9,10,11,12,13,14,15);
}

// =====================================================================
// Kernel 1: flash attention.  grid = (SEQ/64, N*HEADS), block = 256.
// out (workspace): f16 [N, SEQ, EMBED] with head-major columns h*64+d.
// =====================================================================
__global__ __launch_bounds__(256)
void attn_flash_kernel(const float* __restrict__ values,
                       const float* __restrict__ keys,
                       const float* __restrict__ query,
                       const int*   __restrict__ mask,
                       _Float16*    __restrict__ attn_out) {
  __shared__ _Float16 Qs[64 * PIT];
  __shared__ _Float16 Ks[64 * PIT];
  __shared__ _Float16 VsT[64 * PIT];   // transposed: [d][k]
  __shared__ _Float16 Ps[64 * PIT];
  __shared__ int      Ms[64 * 64];
  __shared__ float    pmax[64 * 2], psum[64 * 2];
  __shared__ float    mrun[64], lsum[64], alphaS[64];

  const int qt = blockIdx.x;           // 64-row query tile
  const int nh = blockIdx.y;
  const int n  = nh >> 3;
  const int h  = nh & 7;
  const int q0 = qt * 64;

  const int tid  = threadIdx.x;
  const int wid  = tid >> 5;
  const int lane = tid & 31;
  const int hl   = lane >> 4;
  const int n16  = lane & 15;
  const int r    = wid & 3;            // q sub-block (16 rows)
  const int cg   = wid >> 2;           // column group (32 of 64 cols)

  if (tid < 64) { mrun[tid] = -3.0e38f; lsum[tid] = 0.0f; }

  const float* qbase = query  + ((size_t)(n * SEQ + q0)) * EMBED + h * HD;
  const float* kbase = keys   + ((size_t)n * SEQ) * EMBED + h * HD;
  const float* vbase = values + ((size_t)n * SEQ) * EMBED + h * HD;
  const int*   mbase = mask + (size_t)n * SEQ * SEQ + (size_t)q0 * SEQ;

  // stage Q tile (fp32 -> f16)
#pragma unroll
  for (int i = 0; i < 4; ++i) {
    int idx = tid + i * 256;            // 0..1023
    int row = idx >> 4, seg = idx & 15;
    float4 f = *(const float4*)(qbase + (size_t)row * EMBED + seg * 4);
    _Float16* d = &Qs[row * PIT + seg * 4];
    d[0] = (_Float16)f.x; d[1] = (_Float16)f.y;
    d[2] = (_Float16)f.z; d[3] = (_Float16)f.w;
  }

  v8f o0 = {}; v8f o1 = {};

  for (int kt = 0; kt < SEQ / 64; ++kt) {
    __syncthreads();                    // prior iter done with LDS
    const int k0 = kt * 64;

#if __has_builtin(__builtin_amdgcn_tensor_load_to_lds)
    // ---- async mask tile via Tensor Data Mover (wave 0 issues) ----
    if (tid == 0) {
      unsigned long long ga = (unsigned long long)(uintptr_t)(mbase + k0);
      unsigned ldsa = (unsigned)(uintptr_t)(&Ms[0]);
      u32x4 g0;
      g0.x = 1u;                                    // count=1, user mode
      g0.y = ldsa;                                  // lds_addr
      g0.z = (unsigned)(ga & 0xFFFFFFFFu);          // global_addr[31:0]
      g0.w = (unsigned)((ga >> 32) & 0x01FFFFFFu) | (2u << 30);  // type=2
      i32x8 g1;
      g1[0] = (int)(2u << 16);          // data_size=4B, no multicast
      g1[1] = (int)((unsigned)SEQ << 16);   // tensor_dim0[15:0] @bits63:48
      g1[2] = (int)((unsigned)SEQ << 16);   // tensor_dim1[15:0] @bits95:80
      g1[3] = (int)(64u << 16);         // tile_dim0=64 @bits127:112
      g1[4] = 64;                       // tile_dim1=64 @bits143:128
      g1[5] = SEQ;                      // tensor_dim0_stride low32
      g1[6] = 0; g1[7] = 0;
      i32x4 g2 = {0, 0, 0, 0};
      i32x4 g3 = {0, 0, 0, 0};
      i32x8 g4 = {0, 0, 0, 0, 0, 0, 0, 0};
      __builtin_amdgcn_tensor_load_to_lds(g0, g1, g2, g3, g4, 0);
    }
#else
#pragma unroll
    for (int i = 0; i < 16; ++i) {
      int idx = tid + i * 256;          // 0..4095
      Ms[idx] = mbase[(size_t)(idx >> 6) * SEQ + k0 + (idx & 63)];
    }
#endif

    // ---- stage K and V tiles (overlaps with TDM) ----
#pragma unroll
    for (int i = 0; i < 4; ++i) {
      int idx = tid + i * 256;
      int row = idx >> 4, seg = idx & 15;
      float4 kf = *(const float4*)(kbase + (size_t)(k0 + row) * EMBED + seg * 4);
      _Float16* kd = &Ks[row * PIT + seg * 4];
      kd[0] = (_Float16)kf.x; kd[1] = (_Float16)kf.y;
      kd[2] = (_Float16)kf.z; kd[3] = (_Float16)kf.w;
      float4 vf = *(const float4*)(vbase + (size_t)(k0 + row) * EMBED + seg * 4);
      int d0 = seg * 4;
      VsT[(d0 + 0) * PIT + row] = (_Float16)vf.x;
      VsT[(d0 + 1) * PIT + row] = (_Float16)vf.y;
      VsT[(d0 + 2) * PIT + row] = (_Float16)vf.z;
      VsT[(d0 + 3) * PIT + row] = (_Float16)vf.w;
    }
    if (kt + 1 < SEQ / 64)
      __builtin_prefetch(kbase + (size_t)(k0 + 64) * EMBED, 0, 0);

#if __has_builtin(__builtin_amdgcn_tensor_load_to_lds)
    if (tid == 0) __builtin_amdgcn_s_wait_tensorcnt(0);
#endif
    __syncthreads();

    // ---- S = Q K^T  (each wave: 16x16 tiles (r, 2cg) and (r, 2cg+1)) ----
    const _Float16* arow = &Qs[(r * 16 + n16) * PIT];
    v16h a0 = fragA(arow, hl, 0);
    v16h a1 = fragA(arow, hl, 32);
    v8f s0 = {}; v8f s1 = {};
    {
      const _Float16* b0p = &Ks[((2 * cg + 0) * 16 + n16) * PIT];
      s0 = wmma_f16(a0, fragB(b0p, hl, 0),  s0);
      s0 = wmma_f16(a1, fragB(b0p, hl, 32), s0);
      const _Float16* b1p = &Ks[((2 * cg + 1) * 16 + n16) * PIT];
      s1 = wmma_f16(a0, fragB(b1p, hl, 0),  s1);
      s1 = wmma_f16(a1, fragB(b1p, hl, 32), s1);
    }

    // ---- mask + scale + partial row-max over this wave's 32 columns ----
    const float scale = 0.125f;  // 1/sqrt(64)
#pragma unroll
    for (int v = 0; v < 8; ++v) {
      int rl = r * 16 + v + 8 * hl;     // row in 64-tile (C/D layout)
      int m0 = Ms[rl * 64 + (2 * cg + 0) * 16 + n16];
      int m1 = Ms[rl * 64 + (2 * cg + 1) * 16 + n16];
      float e0 = (m0 == 0) ? -1.0e20f : s0[v] * scale;
      float e1 = (m1 == 0) ? -1.0e20f : s1[v] * scale;
      s0[v] = e0; s1[v] = e1;
      float mx = fmaxf(e0, e1);
      mx = fmaxf(mx, __shfl_xor(mx, 8));
      mx = fmaxf(mx, __shfl_xor(mx, 4));
      mx = fmaxf(mx, __shfl_xor(mx, 2));
      mx = fmaxf(mx, __shfl_xor(mx, 1));
      if (n16 == 0) pmax[rl * 2 + cg] = mx;
    }
    __syncthreads();

    if (tid < 64) {
      float mt = fmaxf(pmax[tid * 2], pmax[tid * 2 + 1]);
      float mo = mrun[tid];
      float mn = fmaxf(mo, mt);
      float al = __expf(mo - mn);
      mrun[tid] = mn; alphaS[tid] = al; lsum[tid] *= al;
    }
    __syncthreads();

    // ---- P = exp(S - m), row sums, rescale O, stash P(f16) ----
#pragma unroll
    for (int v = 0; v < 8; ++v) {
      int rl = r * 16 + v + 8 * hl;
      float mn = mrun[rl];
      float p0 = __expf(s0[v] - mn);
      float p1 = __expf(s1[v] - mn);
      Ps[rl * PIT + (2 * cg + 0) * 16 + n16] = (_Float16)p0;
      Ps[rl * PIT + (2 * cg + 1) * 16 + n16] = (_Float16)p1;
      float sm = p0 + p1;
      sm += __shfl_xor(sm, 8);
      sm += __shfl_xor(sm, 4);
      sm += __shfl_xor(sm, 2);
      sm += __shfl_xor(sm, 1);
      if (n16 == 0) psum[rl * 2 + cg] = sm;
      float al = alphaS[rl];
      o0[v] *= al; o1[v] *= al;
    }
    __syncthreads();
    if (tid < 64) lsum[tid] += psum[tid * 2] + psum[tid * 2 + 1];

    // ---- O += P V  (A = P rows r*16.., B = V^T cols) ----
    const _Float16* prow = &Ps[(r * 16 + n16) * PIT];
    v16h pa0 = fragA(prow, hl, 0);
    v16h pa1 = fragA(prow, hl, 32);
    {
      const _Float16* v0p = &VsT[((2 * cg + 0) * 16 + n16) * PIT];
      o0 = wmma_f16(pa0, fragB(v0p, hl, 0),  o0);
      o0 = wmma_f16(pa1, fragB(v0p, hl, 32), o0);
      const _Float16* v1p = &VsT[((2 * cg + 1) * 16 + n16) * PIT];
      o1 = wmma_f16(pa0, fragB(v1p, hl, 0),  o1);
      o1 = wmma_f16(pa1, fragB(v1p, hl, 32), o1);
    }
  }
  __syncthreads();

  // ---- normalize and write attention output (f16) to workspace ----
  _Float16* obase = attn_out + ((size_t)(n * SEQ + q0)) * EMBED + h * HD;
#pragma unroll
  for (int v = 0; v < 8; ++v) {
    int rl = r * 16 + v + 8 * hl;
    float inv = 1.0f / lsum[rl];
    obase[(size_t)rl * EMBED + (2 * cg + 0) * 16 + n16] = (_Float16)(o0[v] * inv);
    obase[(size_t)rl * EMBED + (2 * cg + 1) * 16 + n16] = (_Float16)(o1[v] * inv);
  }
}

// =====================================================================
// Kernel 2: out = attn(f16) @ W^T + b.  grid=(8192/64, 512/64), block=256.
// =====================================================================
__global__ __launch_bounds__(256)
void fc_out_kernel(const _Float16* __restrict__ A,
                   const float*    __restrict__ W,
                   const float*    __restrict__ bias,
                   float*          __restrict__ out) {
  __shared__ _Float16 As[64 * PIT];
  __shared__ _Float16 Wt[64 * PIT];    // Wt[j_local][k_local] (B is W[j][k])

  const int it = blockIdx.x, jt = blockIdx.y;
  const int tid = threadIdx.x;
  const int wid = tid >> 5, lane = tid & 31;
  const int hl = lane >> 4, n16 = lane & 15;
  const int r = wid & 3, cg = wid >> 2;

  v8f c0 = {}; v8f c1 = {};

  for (int kt = 0; kt < EMBED / 64; ++kt) {
    __syncthreads();
#pragma unroll
    for (int i = 0; i < 2; ++i) {       // stage A (f16 direct copy)
      int idx = tid + i * 256;          // 0..511
      int row = idx >> 3, seg = idx & 7;
      uint2 d = *(const uint2*)(A + ((size_t)(it * 64 + row)) * EMBED +
                                kt * 64 + seg * 8);
      *(uint2*)&As[row * PIT + seg * 8] = d;
    }
#pragma unroll
    for (int i = 0; i < 4; ++i) {       // stage W (f32 -> f16)
      int idx = tid + i * 256;          // 0..1023
      int row = idx >> 4, seg = idx & 15;
      float4 f = *(const float4*)(W + ((size_t)(jt * 64 + row)) * EMBED +
                                  kt * 64 + seg * 4);
      _Float16* d = &Wt[row * PIT + seg * 4];
      d[0] = (_Float16)f.x; d[1] = (_Float16)f.y;
      d[2] = (_Float16)f.z; d[3] = (_Float16)f.w;
    }
    __syncthreads();

    const _Float16* ar = &As[(r * 16 + n16) * PIT];
    v16h a0 = fragA(ar, hl, 0);
    v16h a1 = fragA(ar, hl, 32);
    const _Float16* b0r = &Wt[((2 * cg + 0) * 16 + n16) * PIT];
    c0 = wmma_f16(a0, fragB(b0r, hl, 0),  c0);
    c0 = wmma_f16(a1, fragB(b0r, hl, 32), c0);
    const _Float16* b1r = &Wt[((2 * cg + 1) * 16 + n16) * PIT];
    c1 = wmma_f16(a0, fragB(b1r, hl, 0),  c1);
    c1 = wmma_f16(a1, fragB(b1r, hl, 32), c1);
  }

  const int col0 = jt * 64 + (2 * cg + 0) * 16 + n16;
  const int col1 = jt * 64 + (2 * cg + 1) * 16 + n16;
  const float b0 = bias[col0], b1 = bias[col1];
#pragma unroll
  for (int v = 0; v < 8; ++v) {
    int row = it * 64 + r * 16 + v + 8 * hl;
    out[(size_t)row * EMBED + col0] = c0[v] + b0;
    out[(size_t)row * EMBED + col1] = c1[v] + b1;
  }
}

// =====================================================================
extern "C" void kernel_launch(void* const* d_in, const int* in_sizes, int n_in,
                              void* d_out, int out_size, void* d_ws,
                              size_t ws_size, hipStream_t stream) {
  (void)in_sizes; (void)n_in; (void)out_size; (void)ws_size;
  const float* values = (const float*)d_in[0];
  const float* keys   = (const float*)d_in[1];
  const float* query  = (const float*)d_in[2];
  const int*   mask   = (const int*)d_in[3];
  const float* fc_w   = (const float*)d_in[4];
  const float* fc_b   = (const float*)d_in[5];
  _Float16* attn = (_Float16*)d_ws;            // [N, SEQ, EMBED] f16 = 8 MB

  attn_flash_kernel<<<dim3(SEQ / 64, NB * HEADS), 256, 0, stream>>>(
      values, keys, query, mask, attn);
  fc_out_kernel<<<dim3((NB * SEQ) / 64, EMBED / 64), 256, 0, stream>>>(
      attn, fc_w, fc_b, (float*)d_out);
}
